// MultiHeadSelfAttention_22969485099252
// MI455X (gfx1250) — compile-verified, matrix operand order checked by reading, not run on hardware
//
#include <hip/hip_runtime.h>

#define BS    4
#define SEQ   2048
#define DIMSZ 1024
#define NH    16
#define HD    64
#define MROWS (BS*SEQ)   // 8192

typedef __attribute__((ext_vector_type(16))) __bf16         v16bf;
typedef __attribute__((ext_vector_type(8)))  float          v8f;
typedef __attribute__((ext_vector_type(8)))  unsigned short us8;
typedef __attribute__((ext_vector_type(4)))  unsigned short us4;

union FragBF { v16bf bf; us8 u[2]; };

__device__ __forceinline__ unsigned short f2bf(float f) {
  unsigned int u = __float_as_uint(f);
  u += 0x7FFFu + ((u >> 16) & 1u);      // round-to-nearest-even
  return (unsigned short)(u >> 16);
}

__device__ __forceinline__ v8f wmma_bf16(FragBF a, FragBF b, v8f c) {
  return __builtin_amdgcn_wmma_f32_16x16x32_bf16(
      false, a.bf, false, b.bf, (short)0, c, false, false);
}

// 16x32 bf16 fragment (A layout; B-as-NxK uses the same pattern).
// lane<16: row=lane,     k chunks {0..7},{16..23}
// lane>=16: row=lane-16, k chunks {8..15},{24..31}
__device__ __forceinline__ FragBF load_frag(const unsigned short* __restrict__ base,
                                            int ld, int row, int k0, int half) {
  FragBF f;
  const unsigned short* p = base + (size_t)row * ld + k0;
  f.u[0] = *(const us8*)(p + 8 * half);
  f.u[1] = *(const us8*)(p + 16 + 8 * half);
  return f;
}

// ---------------- f32 -> bf16 convert (vectorized x4) ----------------
__global__ __launch_bounds__(256) void cvt_bf16(const float* __restrict__ src,
                                                unsigned short* __restrict__ dst,
                                                int n) {
  int i = (blockIdx.x * 256 + threadIdx.x) * 4;
  if (i + 3 < n) {
    float4 f = *(const float4*)(src + i);
    us4 r;
    r.x = f2bf(f.x); r.y = f2bf(f.y); r.z = f2bf(f.z); r.w = f2bf(f.w);
    *(us4*)(dst + i) = r;
  }
}

// ---------------- GEMM: Y = X @ W^T + b ----------------
// Wave tile 32x64 (2 A-frags x 4 B-frags, 8 accumulators), register
// double-buffered over K so loads for step k+1 are in flight during the
// 8-WMMA batch of step k (breaks the s_wait_loadcnt 0 serialization).
// mode 0: Q -> scale 1/8, bf16 (b,h,s,d)
// mode 1: K ->            bf16 (b,h,s,d)
// mode 2: V ->            bf16 (b,h,d,s)
// mode 3: out-proj ->     f32  (r, c) + bias
__device__ __forceinline__ void load_a2(FragBF a[2], const unsigned short* __restrict__ X,
                                        int rowBase, int l16, int half, int k0) {
  a[0] = load_frag(X, DIMSZ, rowBase + l16,      k0, half);
  a[1] = load_frag(X, DIMSZ, rowBase + 16 + l16, k0, half);
}
__device__ __forceinline__ void load_b4(FragBF bb[4], const unsigned short* __restrict__ W,
                                        int colBase, int l16, int half, int k0) {
  #pragma unroll
  for (int t = 0; t < 4; t++)
    bb[t] = load_frag(W, DIMSZ, colBase + t * 16 + l16, k0, half);
}
__device__ __forceinline__ void mma8(v8f acc[2][4], FragBF a[2], FragBF bb[4]) {
  #pragma unroll
  for (int mi = 0; mi < 2; mi++)
    #pragma unroll
    for (int t = 0; t < 4; t++)
      acc[mi][t] = wmma_bf16(a[mi], bb[t], acc[mi][t]);
}

__global__ __launch_bounds__(256) void gemm_bf16(const unsigned short* __restrict__ X,
                                                 const unsigned short* __restrict__ W,
                                                 const float* __restrict__ bias,
                                                 void* __restrict__ outp,
                                                 int mode) {
  const int lane  = threadIdx.x & 31;
  const int wave  = threadIdx.x >> 5;
  const int half  = lane >> 4;
  const int l16   = lane & 15;
  const int waveM = wave & 1;    // 2 waves down M
  const int waveN = wave >> 1;   // 4 waves across N
  const int rowBase = blockIdx.y * 64  + waveM * 32;
  const int colBase = blockIdx.x * 256 + waveN * 64;

  v8f vzero = {};
  v8f acc[2][4];
  #pragma unroll
  for (int mi = 0; mi < 2; mi++)
    #pragma unroll
    for (int t = 0; t < 4; t++) acc[mi][t] = vzero;

  FragBF a0[2], b0[4], a1[2], b1[4];
  load_a2(a0, X, rowBase, l16, half, 0);
  load_b4(b0, W, colBase, l16, half, 0);

  for (int k0 = 0; k0 < DIMSZ; k0 += 64) {
    load_a2(a1, X, rowBase, l16, half, k0 + 32);
    load_b4(b1, W, colBase, l16, half, k0 + 32);
    mma8(acc, a0, b0);
    if (k0 + 64 < DIMSZ) {
      load_a2(a0, X, rowBase, l16, half, k0 + 64);
      load_b4(b0, W, colBase, l16, half, k0 + 64);
    }
    mma8(acc, a1, b1);
  }

  const float scale = (mode == 0) ? 0.125f : 1.0f;  // 1/sqrt(64)
  #pragma unroll
  for (int mi = 0; mi < 2; mi++) {
    #pragma unroll
    for (int t = 0; t < 4; t++) {
      const int   c  = colBase + t * 16 + l16;
      const float bv = bias[c];
      const int   h  = c >> 6, d = c & 63;
      #pragma unroll
      for (int j = 0; j < 8; j++) {
        const int r = rowBase + mi * 16 + j + 8 * half;  // 0..8191
        const float v = acc[mi][t][j] + bv;
        if (mode == 3) {
          ((float*)outp)[(size_t)r * DIMSZ + c] = v;
        } else {
          const int b = r >> 11, s = r & (SEQ - 1);
          size_t idx;
          if (mode == 2) idx = ((size_t)(b * NH + h) * HD + d) * SEQ + s;
          else           idx = ((size_t)(b * NH + h) * SEQ + s) * HD + d;
          ((unsigned short*)outp)[idx] = f2bf(v * scale);
        }
      }
    }
  }
}

// ---------------- Flash attention ----------------
// Qh,Kh: (b,h,s,d) bf16 (Q pre-scaled); Vt: (b,h,d,s) bf16; ctx: (b,s,DIM) bf16
// K/V chunks (32 keys) are staged into LDS with GLOBAL_LOAD_ASYNC_TO_LDS_B128
// (one 512B op per wave per matrix), double-buffered across the key loop.
__global__ __launch_bounds__(256) void attn_fwd(const unsigned short* __restrict__ Qh,
                                                const unsigned short* __restrict__ Kh,
                                                const unsigned short* __restrict__ Vt,
                                                const int* __restrict__ mask,
                                                unsigned short* __restrict__ ctx) {
  __shared__ alignas(16) unsigned short kc[2][32 * 64];   // 2 x 4KB
  __shared__ alignas(16) unsigned short vc[2][64 * 32];   // 2 x 4KB
  __shared__ alignas(16) unsigned short pbuf[8][16 * 32]; // per-wave P tile

  const int lane = threadIdx.x & 31;
  const int wave = threadIdx.x >> 5;
  const int half = lane >> 4;
  const int l16  = lane & 15;
  const int bh   = blockIdx.y;          // 0..63
  const int b    = bh >> 4;
  const int h    = bh & 15;
  const int qbase = blockIdx.x * 128 + wave * 16;

  const unsigned short* Qp = Qh + (size_t)bh * SEQ * HD;
  const unsigned short* Kp = Kh + (size_t)bh * SEQ * HD;
  const unsigned short* Vp = Vt + (size_t)bh * HD * SEQ;
  const int* mrow = mask + b * SEQ;

  // LDS byte offsets for async destinations
  const unsigned kc_base = (unsigned)(size_t)&kc[0][0];
  const unsigned vc_base = (unsigned)(size_t)&vc[0][0];
  // K: this wave covers bytes [wave*512, wave*512+512) of the contiguous 4KB chunk
  const unsigned k_off = (unsigned)(wave * 512 + lane * 16);
  // V: wave covers rows [wave*8, wave*8+8); lane -> (row, 16B quarter)
  const int      vrow  = wave * 8 + (lane >> 2);
  const unsigned vq    = (unsigned)((lane & 3) * 16);
  const unsigned v_off = (unsigned)(vrow * 64) + vq;

  auto issue_chunk = [&](int k0, int buf) {
    const char* gk = (const char*)Kp + (size_t)k0 * (HD * 2) + k_off;
    unsigned    lk = kc_base + (unsigned)(buf * 4096) + k_off;
    asm volatile("global_load_async_to_lds_b128 %0, %1, off"
                 :: "v"(lk), "v"(gk) : "memory");
    const char* gv = (const char*)Vp + (size_t)vrow * (SEQ * 2) + (size_t)k0 * 2 + vq;
    unsigned    lv = vc_base + (unsigned)(buf * 4096) + v_off;
    asm volatile("global_load_async_to_lds_b128 %0, %1, off"
                 :: "v"(lv), "v"(gv) : "memory");
  };

  // Q fragments resident in VGPRs for the whole key loop
  FragBF aq0 = load_frag(Qp, HD, qbase + l16, 0,  half);
  FragBF aq1 = load_frag(Qp, HD, qbase + l16, 32, half);

  v8f vzero = {};
  v8f ctxa[4];
  #pragma unroll
  for (int t = 0; t < 4; t++) ctxa[t] = vzero;

  float m_i[8], l_i[8];
  #pragma unroll
  for (int j = 0; j < 8; j++) { m_i[j] = -__builtin_inff(); l_i[j] = 0.f; }

  unsigned short* pb = pbuf[wave];

  issue_chunk(0, 0);

  for (int ci = 0; ci < SEQ / 32; ci++) {
    const int k0 = ci * 32;
    // wait for our own async writes into buf (ci&1); barrier publishes all
    // waves' writes and guarantees nobody still reads buf ((ci+1)&1)
    asm volatile("s_wait_asynccnt 0x0" ::: "memory");
    __syncthreads();
    if (ci + 1 < SEQ / 32) issue_chunk(k0 + 32, (ci + 1) & 1);

    const unsigned short* Kc = kc[ci & 1];
    const unsigned short* Vc = vc[ci & 1];

    // ---- scores chunk: S(16x32) = Q(16x64) @ K_chunk^T ----
    v8f s0 = vzero, s1 = vzero;
    {
      FragBF bk;
      bk = load_frag(Kc, HD, l16,      0,  half); s0 = wmma_bf16(aq0, bk, s0);
      bk = load_frag(Kc, HD, l16,      32, half); s0 = wmma_bf16(aq1, bk, s0);
      bk = load_frag(Kc, HD, 16 + l16, 0,  half); s1 = wmma_bf16(aq0, bk, s1);
      bk = load_frag(Kc, HD, 16 + l16, 32, half); s1 = wmma_bf16(aq1, bk, s1);
    }

    // ---- padding mask (column = this lane's N) ----
    if (mrow[k0 + l16] == 0) {
      #pragma unroll
      for (int j = 0; j < 8; j++) s0[j] = -__builtin_inff();
    }
    if (mrow[k0 + 16 + l16] == 0) {
      #pragma unroll
      for (int j = 0; j < 8; j++) s1[j] = -__builtin_inff();
    }

    // ---- chunk row-max across the 16-lane half-groups ----
    float cm[8];
    #pragma unroll
    for (int j = 0; j < 8; j++) cm[j] = fmaxf(s0[j], s1[j]);
    #pragma unroll
    for (int off = 8; off >= 1; off >>= 1) {
      #pragma unroll
      for (int j = 0; j < 8; j++) cm[j] = fmaxf(cm[j], __shfl_xor(cm[j], off, 32));
    }

    // ---- online softmax update ----
    float p0[8], p1[8], rs[8];
    #pragma unroll
    for (int j = 0; j < 8; j++) {
      const float mn    = fmaxf(m_i[j], cm[j]);
      const float alpha = (m_i[j] == -__builtin_inff()) ? 0.f : __expf(m_i[j] - mn);
      p0[j] = (s0[j] == -__builtin_inff()) ? 0.f : __expf(s0[j] - mn);
      p1[j] = (s1[j] == -__builtin_inff()) ? 0.f : __expf(s1[j] - mn);
      m_i[j] = mn;
      l_i[j] *= alpha;
      #pragma unroll
      for (int t = 0; t < 4; t++) ctxa[t][j] *= alpha;
      rs[j] = p0[j] + p1[j];
    }
    #pragma unroll
    for (int off = 8; off >= 1; off >>= 1) {
      #pragma unroll
      for (int j = 0; j < 8; j++) rs[j] += __shfl_xor(rs[j], off, 32);
    }
    #pragma unroll
    for (int j = 0; j < 8; j++) l_i[j] += rs[j];

    // ---- C-layout -> A-layout via LDS (same-wave DS ops are in-order) ----
    #pragma unroll
    for (int j = 0; j < 8; j++) {
      const int r = j + 8 * half;
      pb[r * 32 + l16]      = f2bf(p0[j]);
      pb[r * 32 + 16 + l16] = f2bf(p1[j]);
    }
    FragBF pf;
    {
      const unsigned short* pr = pb + l16 * 32;
      pf.u[0] = *(const us8*)(pr + 8 * half);
      pf.u[1] = *(const us8*)(pr + 16 + 8 * half);
    }

    // ---- ctx(16x64) += P(16x32) @ V_chunk(32x64) ----
    #pragma unroll
    for (int t = 0; t < 4; t++) {
      FragBF bv = load_frag(Vc, 32, t * 16 + l16, 0, half);
      ctxa[t] = wmma_bf16(pf, bv, ctxa[t]);
    }
  }

  // ---- epilogue: normalize and store (b,s,DIM) ----
  float inv[8];
  #pragma unroll
  for (int j = 0; j < 8; j++) inv[j] = (l_i[j] > 0.f) ? (1.f / l_i[j]) : 0.f;
  #pragma unroll
  for (int t = 0; t < 4; t++) {
    #pragma unroll
    for (int j = 0; j < 8; j++) {
      const int srow = qbase + j + 8 * half;
      const size_t idx = ((size_t)(b * SEQ + srow)) * DIMSZ + h * HD + t * 16 + l16;
      ctx[idx] = f2bf(ctxa[t][j] * inv[j]);
    }
  }
}

// ---------------- host ----------------
extern "C" void kernel_launch(void* const* d_in, const int* in_sizes, int n_in,
                              void* d_out, int out_size, void* d_ws, size_t ws_size,
                              hipStream_t stream) {
  const float* q    = (const float*)d_in[0];
  const float* k    = (const float*)d_in[1];
  const float* v    = (const float*)d_in[2];
  const int*   mask = (const int*)d_in[3];
  const float* Wq   = (const float*)d_in[4];
  const float* bq   = (const float*)d_in[5];
  const float* Wk   = (const float*)d_in[6];
  const float* bk   = (const float*)d_in[7];
  const float* Wv   = (const float*)d_in[8];
  const float* bv   = (const float*)d_in[9];
  const float* Wo   = (const float*)d_in[10];
  const float* bo   = (const float*)d_in[11];

  char* ws = (char*)d_ws;
  const size_t MB = 1ull << 20;
  unsigned short* qbf  = (unsigned short*)(ws + 0   * MB);  // 16 MB
  unsigned short* kbf  = (unsigned short*)(ws + 16  * MB);  // 16 MB
  unsigned short* vbf  = (unsigned short*)(ws + 32  * MB);  // 16 MB
  unsigned short* wqb  = (unsigned short*)(ws + 48  * MB);  //  2 MB
  unsigned short* wkb  = (unsigned short*)(ws + 50  * MB);
  unsigned short* wvb  = (unsigned short*)(ws + 52  * MB);
  unsigned short* wob  = (unsigned short*)(ws + 54  * MB);
  unsigned short* Qhp  = (unsigned short*)(ws + 56  * MB);  // 16 MB (b,h,s,d)
  unsigned short* Khp  = (unsigned short*)(ws + 72  * MB);  // 16 MB (b,h,s,d)
  unsigned short* Vtp  = (unsigned short*)(ws + 88  * MB);  // 16 MB (b,h,d,s)
  unsigned short* ctxp = (unsigned short*)(ws + 104 * MB);  // 16 MB (b,s,DIM)

  const int nX = MROWS * DIMSZ;   // 8,388,608
  const int nW = DIMSZ * DIMSZ;   // 1,048,576

  cvt_bf16<<<nX / 1024, 256, 0, stream>>>(q,  qbf, nX);
  cvt_bf16<<<nX / 1024, 256, 0, stream>>>(k,  kbf, nX);
  cvt_bf16<<<nX / 1024, 256, 0, stream>>>(v,  vbf, nX);
  cvt_bf16<<<nW / 1024, 256, 0, stream>>>(Wq, wqb, nW);
  cvt_bf16<<<nW / 1024, 256, 0, stream>>>(Wk, wkb, nW);
  cvt_bf16<<<nW / 1024, 256, 0, stream>>>(Wv, wvb, nW);
  cvt_bf16<<<nW / 1024, 256, 0, stream>>>(Wo, wob, nW);

  dim3 gGemm(DIMSZ / 256, MROWS / 64);  // (4,128)
  gemm_bf16<<<gGemm, 256, 0, stream>>>(qbf, wqb, bq, Qhp, 0);
  gemm_bf16<<<gGemm, 256, 0, stream>>>(kbf, wkb, bk, Khp, 1);
  gemm_bf16<<<gGemm, 256, 0, stream>>>(vbf, wvb, bv, Vtp, 2);

  attn_fwd<<<dim3(SEQ / 128, BS * NH), 256, 0, stream>>>(Qhp, Khp, Vtp, mask, ctxp);

  gemm_bf16<<<gGemm, 256, 0, stream>>>(ctxp, wob, bo, (float*)d_out, 3);
}
